// AsyncLongContextAttention_32186484916551
// MI455X (gfx1250) — matrix operand order repaired and checked by reading, not compiled, not run on hardware
//
#include <hip/hip_runtime.h>
#include <hip/hip_bf16.h>

typedef __attribute__((ext_vector_type(16))) _Float16 v16h;
typedef __attribute__((ext_vector_type(8)))  _Float16 v8h;
typedef __attribute__((ext_vector_type(4)))  _Float16 v4h;
typedef __attribute__((ext_vector_type(2)))  _Float16 v2h;
typedef __attribute__((ext_vector_type(8)))  float    v8f;
typedef __attribute__((ext_vector_type(4)))  float    v4f;
typedef __attribute__((ext_vector_type(8)))  int      v8i_;

namespace {
constexpr int kB  = 2;     // batch
constexpr int kS  = 2048;  // sequence
constexpr int kH  = 16;    // heads
constexpr int kD  = 128;   // head size
constexpr int kBR = 128;          // query rows per block (8 waves x 16)
constexpr int kBC = 64;           // key rows per inner tile (4 x 16-key groups)
constexpr int kNT = kS / kBC;     // 32 inner tiles
constexpr int kWaves = kBR / 16;  // 8 waves, 256 threads
constexpr int kVStride = 72;      // padded halves per transposed-V row (144B = 9*16B)
constexpr float kScale = 0.08838834764831845f;   // 1/sqrt(128)
constexpr float kLog2e = 1.4426950408889634f;
}

__device__ __forceinline__ v8f zero8() {
  v8f z;
#pragma unroll
  for (int i = 0; i < 8; ++i) z[i] = 0.0f;
  return z;
}

__device__ __forceinline__ v16h cat8(v8h lo, v8h hi) {
  return __builtin_shufflevector(lo, hi, 0,1,2,3,4,5,6,7,8,9,10,11,12,13,14,15);
}

union PB { v8i_ i; v16h h; };

__global__ __launch_bounds__(kWaves * 32, 1)
void fa_fwd_f16wmma_kernel(const float* __restrict__ Q,
                           const float* __restrict__ K,
                           const float* __restrict__ V,
                           float* __restrict__ O)
{
  // Double-buffered LDS: K row-major f16 [key][d]; V transposed f16 [d][key+pad].
  __shared__ _Float16 sK[2][kBC * kD];        // 2 x 16 KB
  __shared__ _Float16 sV[2][kD * kVStride];   // 2 x 18 KB

  const int tid  = threadIdx.x;
  const int wave = tid >> 5;
  const int lane = tid & 31;
  const int g    = lane >> 4;   // half-wave id (0/1)
  const int ln   = lane & 15;

  const int bh = blockIdx.y;
  const int b  = bh >> 4;       // / kH
  const int h  = bh & 15;       // % kH
  const int q0 = blockIdx.x * kBR + wave * 16; // wave's 16-row Q strip

  const size_t rowstride = (size_t)kH * kD;
  const float* kbh = K + (((size_t)b * kS) * kH + h) * kD;
  const float* vbh = V + (((size_t)b * kS) * kH + h) * kD;

  // Q^T B-fragments: lane holds q-col ln; halves are 16 contiguous d = 32c+16g+i.
  v16h qf[4];
  {
    const float* qr = Q + (((size_t)b * kS + (q0 + ln)) * kH + h) * kD;
#pragma unroll
    for (int c = 0; c < 4; ++c) {
      const int d0 = 32 * c + 16 * g;
      v4f x0 = *(const v4f*)(qr + d0);
      v4f x1 = *(const v4f*)(qr + d0 + 4);
      v4f x2 = *(const v4f*)(qr + d0 + 8);
      v4f x3 = *(const v4f*)(qr + d0 + 12);
      v16h q;
#pragma unroll
      for (int i = 0; i < 4; ++i) {
        q[i]      = (_Float16)(x0[i] * kScale);
        q[4 + i]  = (_Float16)(x1[i] * kScale);
        q[8 + i]  = (_Float16)(x2[i] * kScale);
        q[12 + i] = (_Float16)(x3[i] * kScale);
      }
      qf[c] = q;
    }
  }

  // O^T accumulators; lane holds q-col ln, reg r -> d = dg*16 + r + 8g.
  v8f o[8];
#pragma unroll
  for (int f = 0; f < 8; ++f) o[f] = zero8();
  float mrun = -1e30f;
  float lrun = 0.0f;

  // Staging ownership: K: thread = (key, 32-d window). V: thread = (4 keys, 8-d window).
  const int skey = tid >> 2;          // 0..63
  const int sdK  = (tid & 3) * 32;    // 0,32,64,96
  const int vkg  = tid >> 4;          // 0..15 -> keys 4*vkg..+3
  const int vd0  = (tid & 15) * 8;    // 0..120

  // ---- stage tile 0 ----
  {
    const float* kr = kbh + (size_t)skey * rowstride + sdK;
    _Float16* kd = &sK[0][skey * kD + sdK];
#pragma unroll
    for (int j = 0; j < 4; ++j) {
      v4f a = *(const v4f*)(kr + 8 * j);
      v4f c = *(const v4f*)(kr + 8 * j + 4);
      v8h hh;
#pragma unroll
      for (int i = 0; i < 4; ++i) { hh[i] = (_Float16)a[i]; hh[4 + i] = (_Float16)c[i]; }
      *(v8h*)(kd + 8 * j) = hh;
    }
    const float* vr = vbh + (size_t)(4 * vkg) * rowstride + vd0;
    v4f vA[4], vB[4];
#pragma unroll
    for (int j = 0; j < 4; ++j) {
      vA[j] = *(const v4f*)(vr + j * rowstride);
      vB[j] = *(const v4f*)(vr + j * rowstride + 4);
    }
#pragma unroll
    for (int i = 0; i < 4; ++i) {
      v4h h0 = {(_Float16)vA[0][i], (_Float16)vA[1][i], (_Float16)vA[2][i], (_Float16)vA[3][i]};
      v4h h1 = {(_Float16)vB[0][i], (_Float16)vB[1][i], (_Float16)vB[2][i], (_Float16)vB[3][i]};
      *(v4h*)(&sV[0][(vd0 + i) * kVStride + 4 * vkg])     = h0;
      *(v4h*)(&sV[0][(vd0 + 4 + i) * kVStride + 4 * vkg]) = h1;
    }
  }
  __syncthreads();

  for (int kt = 0; kt < kNT; ++kt) {
    const int buf  = kt & 1;
    const int nbuf = buf ^ 1;
    const bool more = (kt + 1 < kNT);

    // ---- issue global loads for tile kt+1 (latency covered by QK WMMAs) ----
    v4f kxa[4], kxb[4], vxa[4], vxb[4];
    if (more) {
      const int kb1 = (kt + 1) * kBC;
      const float* kr = kbh + (size_t)(kb1 + skey) * rowstride + sdK;
#pragma unroll
      for (int j = 0; j < 4; ++j) {
        kxa[j] = *(const v4f*)(kr + 8 * j);
        kxb[j] = *(const v4f*)(kr + 8 * j + 4);
      }
      const float* vr = vbh + (size_t)(kb1 + 4 * vkg) * rowstride + vd0;
#pragma unroll
      for (int j = 0; j < 4; ++j) {
        vxa[j] = *(const v4f*)(vr + j * rowstride);
        vxb[j] = *(const v4f*)(vr + j * rowstride + 4);
      }
      if (kt + 2 < kNT) {  // warm L2 for tile kt+2
        __builtin_prefetch(kr + kBC * rowstride, 0, 1);
        __builtin_prefetch(vr + kBC * rowstride, 0, 1);
      }
    }

    // ---- S^T = K * Q^T: 4 key groups x 4 d-chunks = 16 WMMAs ----
    v8f s[4];
#pragma unroll
    for (int grp = 0; grp < 4; ++grp) s[grp] = zero8();
#pragma unroll
    for (int c = 0; c < 4; ++c) {
#pragma unroll
      for (int grp = 0; grp < 4; ++grp) {
        const _Float16* kp = &sK[buf][(grp * 16 + ln) * kD + 32 * c + 16 * g];
        v16h kf = cat8(*(const v8h*)kp, *(const v8h*)(kp + 8));
        s[grp] = __builtin_amdgcn_wmma_f32_16x16x32_f16(false, kf, false, qf[c],
                                                        (short)0, s[grp], false, false);
      }
    }

    // ---- online softmax: in-lane over 32 regs + one xor16 shuffle ----
    float x = s[0][0];
#pragma unroll
    for (int grp = 0; grp < 4; ++grp)
#pragma unroll
      for (int r = 0; r < 8; ++r) x = fmaxf(x, s[grp][r]);
    x = fmaxf(x, __shfl_xor(x, 16, 32));
    const float mnew = fmaxf(mrun, x);
    const float alpha = __builtin_amdgcn_exp2f((mrun - mnew) * kLog2e);
    mrun = mnew;

    float pv[4][8];
    float t = 0.0f;
#pragma unroll
    for (int grp = 0; grp < 4; ++grp)
#pragma unroll
      for (int r = 0; r < 8; ++r) {
        pv[grp][r] = __builtin_amdgcn_exp2f((s[grp][r] - mnew) * kLog2e);
        t += pv[grp][r];
      }
    t += __shfl_xor(t, 16, 32);
    lrun = lrun * alpha + t;

#pragma unroll
    for (int f = 0; f < 8; ++f)
#pragma unroll
      for (int r = 0; r < 8; ++r)
        o[f][r] *= alpha;

    // ---- mid-iteration: convert staged fp32 -> packed f16 (frees 32 VGPRs) ----
    v8h kk[4];
    v4h vh0[4], vh1[4];
    if (more) {
#pragma unroll
      for (int j = 0; j < 4; ++j) {
        v8h hh;
#pragma unroll
        for (int i = 0; i < 4; ++i) { hh[i] = (_Float16)kxa[j][i]; hh[4 + i] = (_Float16)kxb[j][i]; }
        kk[j] = hh;
      }
#pragma unroll
      for (int i = 0; i < 4; ++i) {
        vh0[i] = (v4h){(_Float16)vxa[0][i], (_Float16)vxa[1][i], (_Float16)vxa[2][i], (_Float16)vxa[3][i]};
        vh1[i] = (v4h){(_Float16)vxb[0][i], (_Float16)vxb[1][i], (_Float16)vxb[2][i], (_Float16)vxb[3][i]};
      }
    }

    // ---- build P^T B-fragments (keys 0..31 and 32..63) via xor16 exchange ----
    union HU { v2h h; int u; };
    int own[4][4], oth[4][4];
#pragma unroll
    for (int grp = 0; grp < 4; ++grp)
#pragma unroll
      for (int j = 0; j < 4; ++j) {
        HU u; u.h[0] = (_Float16)pv[grp][2 * j]; u.h[1] = (_Float16)pv[grp][2 * j + 1];
        own[grp][j] = u.u;
      }
#pragma unroll
    for (int grp = 0; grp < 4; ++grp)
#pragma unroll
      for (int j = 0; j < 4; ++j)
        oth[grp][j] = __shfl_xor(own[grp][j], 16, 32);

    PB pbA, pbB;
#pragma unroll
    for (int j = 0; j < 4; ++j) {
      pbA.i[j]     = g ? oth[1][j] : own[0][j];  // keys 16g+0..7
      pbA.i[4 + j] = g ? own[1][j] : oth[0][j];  // keys 16g+8..15
      pbB.i[j]     = g ? oth[3][j] : own[2][j];  // keys 32+16g+0..7
      pbB.i[4 + j] = g ? own[3][j] : oth[2][j];  // keys 32+16g+8..15
    }

    // ---- O^T += V^T * P^T: 8 d-groups x 2 key-halves = 16 WMMAs ----
#pragma unroll
    for (int dg = 0; dg < 8; ++dg) {
      const _Float16* vpA = &sV[buf][(dg * 16 + ln) * kVStride + 16 * g];
      v16h vfA = cat8(*(const v8h*)vpA, *(const v8h*)(vpA + 8));
      o[dg] = __builtin_amdgcn_wmma_f32_16x16x32_f16(false, vfA, false, pbA.h,
                                                     (short)0, o[dg], false, false);
      const _Float16* vpB = vpA + 32;
      v16h vfB = cat8(*(const v8h*)vpB, *(const v8h*)(vpB + 8));
      o[dg] = __builtin_amdgcn_wmma_f32_16x16x32_f16(false, vfB, false, pbB.h,
                                                     (short)0, o[dg], false, false);
    }

    // ---- store staged tile kt+1 into alternate buffers (vectorized DS) ----
    if (more) {
      _Float16* kd = &sK[nbuf][skey * kD + sdK];
#pragma unroll
      for (int j = 0; j < 4; ++j)
        *(v8h*)(kd + 8 * j) = kk[j];
#pragma unroll
      for (int i = 0; i < 4; ++i) {
        *(v4h*)(&sV[nbuf][(vd0 + i) * kVStride + 4 * vkg])     = vh0[i];
        *(v4h*)(&sV[nbuf][(vd0 + 4 + i) * kVStride + 4 * vkg]) = vh1[i];
      }
    }
    __syncthreads();
  }

  // ---- normalize and store O: lane owns q row (q0+ln); d runs are contiguous ----
  const float inv = 1.0f / lrun;
  float* orow = O + (((size_t)b * kS + (q0 + ln)) * kH + h) * kD;
#pragma unroll
  for (int dg = 0; dg < 8; ++dg) {
    v4f t0, t1;
#pragma unroll
    for (int i = 0; i < 4; ++i) { t0[i] = o[dg][i] * inv; t1[i] = o[dg][4 + i] * inv; }
    *(v4f*)(orow + dg * 16 + 8 * g)     = t0;
    *(v4f*)(orow + dg * 16 + 8 * g + 4) = t1;
  }
}

extern "C" void kernel_launch(void* const* d_in, const int* in_sizes, int n_in,
                              void* d_out, int out_size, void* d_ws, size_t ws_size,
                              hipStream_t stream) {
  const float* Q = (const float*)d_in[0];
  const float* K = (const float*)d_in[1];
  const float* V = (const float*)d_in[2];
  float* O = (float*)d_out;
  dim3 grid(kS / kBR, kB * kH);
  dim3 block(kWaves * 32);
  hipLaunchKernelGGL(fa_fwd_f16wmma_kernel, grid, block, 0, stream, Q, K, V, O);
}
